// MultiSingleStreamBlockLoraProcessor_52450140619276
// MI455X (gfx1250) — compile-verified, matrix operand order checked by reading, not run on hardware
//
#include <hip/hip_runtime.h>
#include <hip/hip_bf16.h>
#include <stdint.h>

// ---------------------------------------------------------------- constants
#define DIM   3072
#define HEADS 24
#define HD    128
#define RANK  64
#define COND  1024
#define BLK   2048
#define SEQ   3072   // BLK + COND
#define NIP   64
#define IP_SCALE 1.0f
#define SOFT_SCALE 0.08838834764831845f   // 1/sqrt(128)

// ---------------------------------------------------------------- types
typedef __attribute__((ext_vector_type(16))) __bf16 v16bf;
typedef __attribute__((ext_vector_type(8)))  float  v8f;
typedef __attribute__((ext_vector_type(4)))  unsigned int v4u;
typedef __attribute__((ext_vector_type(8)))  int    v8i;
typedef __attribute__((ext_vector_type(4)))  int    v4i;

#define WMMA_BF16(a, b, c) \
  __builtin_amdgcn_wmma_f32_16x16x32_bf16(false, (a), false, (b), (short)0, (c), false, false)

#define LDS_OFF(p) ((unsigned)(uintptr_t)(p))

#if __has_builtin(__builtin_amdgcn_tensor_load_to_lds)
#define HAVE_TDM 1
#else
#define HAVE_TDM 0
#endif

// ------------------------------------------------------- TDM 2D tile -> LDS
// Builds a D# (group0 + group1) for a 2D bf16 tile and issues TENSOR_LOAD_TO_LDS.
// data_size=1 (2 bytes). tensor dims / stride in elements. OOB rows/cols read 0.
#if HAVE_TDM
__device__ __forceinline__ void tdm_load_2d_bf16(unsigned lds_off, const void* gptr,
                                                 unsigned tensor_d0, unsigned tensor_d1,
                                                 unsigned tile_d0, unsigned tile_d1,
                                                 unsigned long long stride0_elems) {
  unsigned long long ga = (unsigned long long)(uintptr_t)gptr;
  v4u g0;
  g0[0] = 1u;                                              // count=1, user descriptor
  g0[1] = lds_off;                                         // lds_addr (bytes)
  g0[2] = (unsigned)(ga & 0xFFFFFFFFull);                  // global_addr[31:0]
  g0[3] = (unsigned)((ga >> 32) & 0x1FFFFFFull) | (2u << 30); // addr[56:32] | type=2
  v8i g1;
  g1[0] = (int)(1u << 16);                                 // wg_mask=0, data_size=2B
  g1[1] = (int)((tensor_d0 & 0xFFFFu) << 16);              // tensor_dim0 lo
  g1[2] = (int)(((tensor_d0 >> 16) & 0xFFFFu) | ((tensor_d1 & 0xFFFFu) << 16));
  g1[3] = (int)(((tensor_d1 >> 16) & 0xFFFFu) | ((tile_d0 & 0xFFFFu) << 16));
  g1[4] = (int)(tile_d1 & 0xFFFFu);                        // tile_dim1, tile_dim2=0
  g1[5] = (int)(unsigned)(stride0_elems & 0xFFFFFFFFull);  // dim0 stride lo32
  g1[6] = (int)(unsigned)((stride0_elems >> 32) & 0xFFFFull);
  g1[7] = 0;
  v4i gz = {0, 0, 0, 0};
#if defined(__clang_major__) && (__clang_major__ >= 23)
  v8i gz8 = {0, 0, 0, 0, 0, 0, 0, 0};
  __builtin_amdgcn_tensor_load_to_lds(g0, g1, gz, gz, gz8, 0);
#else
  __builtin_amdgcn_tensor_load_to_lds(g0, g1, gz, gz, 0);
#endif
}
#endif

// ---------------------------------------------------------------- cast
__global__ void cast_f32_bf16(const float* __restrict__ s, __bf16* __restrict__ d, long n) {
  long i = (long)blockIdx.x * blockDim.x + threadIdx.x;
  long stride = (long)gridDim.x * blockDim.x;
  for (; i < n; i += stride) d[i] = (__bf16)s[i];
}

// ---------------------------------------------------------------- GEMM
// C[M,N] = A[M,K] (bf16 row major) @ W[N,K]^T (bf16 row major) + bias[N] (+C if accum)
// block = 256 threads = 8 waves. Tile 64(M) x 128(N), K stage = 64, double-buffered
// LDS fed by the Tensor Data Mover: stage i+1 is issued before computing stage i;
// TENSORcnt completes in order, so s_wait_tensorcnt(2) = "current stage landed".
// K must be a multiple of 64 (all call sites: 3072 or 64).
// Fragment k-slot mapping (same for A and B): lane-half h, slot j -> k = 16*h + j,
// so every fragment is one contiguous 32B LDS read.
__global__ void wmma_gemm_nt(const __bf16* __restrict__ A, const __bf16* __restrict__ W,
                             const float* __restrict__ bias, float* __restrict__ C,
                             int M, int N, int K, int accum) {
  __shared__ __bf16 As[2][64 * 64];
  __shared__ __bf16 Bs[2][128 * 64];
  const int tid  = threadIdx.x;
  const int wave = tid >> 5;
  const int lane = tid & 31;
  const int half = lane >> 4;
  const int l16  = lane & 15;
  const int mstrip = wave & 3;
  const int ngrp   = wave >> 2;
  const int m0 = blockIdx.y * 64;
  const int n0 = blockIdx.x * 128;

  v8f acc[4];
  v8f z8 = {0.f, 0.f, 0.f, 0.f, 0.f, 0.f, 0.f, 0.f};
#pragma unroll
  for (int t = 0; t < 4; ++t) acc[t] = z8;

  const int mrows = (M - m0) < 64 ? (M - m0) : 64;
  const int nrows = (N - n0) < 128 ? (N - n0) : 128;
  const int nstage = K >> 6;

#if HAVE_TDM
  if (tid < 32) {  // wave 0 drives the TDM for the whole workgroup
    tdm_load_2d_bf16(LDS_OFF(&As[0][0]), A + (size_t)m0 * K,
                     (unsigned)K, (unsigned)(M - m0), 64u, (unsigned)mrows,
                     (unsigned long long)K);
    tdm_load_2d_bf16(LDS_OFF(&Bs[0][0]), W + (size_t)n0 * K,
                     (unsigned)K, (unsigned)(N - n0), 64u, (unsigned)nrows,
                     (unsigned long long)K);
  }
#endif

  for (int s = 0; s < nstage; ++s) {
    const int cur = s & 1;
    const int k0 = s << 6;
#if HAVE_TDM
    if (tid < 32) {
      if (s + 1 < nstage) {   // prefetch next stage into the other buffer
        tdm_load_2d_bf16(LDS_OFF(&As[1 - cur][0]), A + (size_t)m0 * K + k0 + 64,
                         (unsigned)(K - k0 - 64), (unsigned)(M - m0), 64u, (unsigned)mrows,
                         (unsigned long long)K);
        tdm_load_2d_bf16(LDS_OFF(&Bs[1 - cur][0]), W + (size_t)n0 * K + k0 + 64,
                         (unsigned)(K - k0 - 64), (unsigned)(N - n0), 64u, (unsigned)nrows,
                         (unsigned long long)K);
        __builtin_amdgcn_s_wait_tensorcnt(2);  // oldest pair (stage s) complete
      } else {
        __builtin_amdgcn_s_wait_tensorcnt(0);
      }
    }
#else
    for (int i = tid; i < 64 * 8; i += 256) {
      int r = i >> 3, c = (i & 7) * 8;
      uint4 v = {0u, 0u, 0u, 0u};
      if (r < mrows) v = *(const uint4*)&A[(size_t)(m0 + r) * K + k0 + c];
      *(uint4*)&As[cur][r * 64 + c] = v;
    }
    for (int i = tid; i < 128 * 8; i += 256) {
      int r = i >> 3, c = (i & 7) * 8;
      uint4 v = {0u, 0u, 0u, 0u};
      if (r < nrows) v = *(const uint4*)&W[(size_t)(n0 + r) * K + k0 + c];
      *(uint4*)&Bs[cur][r * 64 + c] = v;
    }
#endif
    __syncthreads();  // stage s visible to all waves

#pragma unroll
    for (int c = 0; c < 2; ++c) {   // two k-chunks of 32
      v16bf afrag = *(const v16bf*)&As[cur][(mstrip * 16 + l16) * 64 + c * 32 + half * 16];
#pragma unroll
      for (int t = 0; t < 4; ++t) {
        v16bf bfrag = *(const v16bf*)&Bs[cur][(ngrp * 64 + t * 16 + l16) * 64 + c * 32 + half * 16];
        acc[t] = WMMA_BF16(afrag, bfrag, acc[t]);
      }
    }
    __syncthreads();  // all reads of buf[cur] done before stage s+2 overwrites it
  }

  // epilogue: C/D layout -> VGPR r holds row (r + 8*half), col = l16 within tile
#pragma unroll
  for (int t = 0; t < 4; ++t) {
    int col = n0 + ngrp * 64 + t * 16 + l16;
    if (col >= N) continue;
    float bv = bias ? bias[col] : 0.f;
#pragma unroll
    for (int r = 0; r < 8; ++r) {
      int row = m0 + mstrip * 16 + r + half * 8;
      if (row < M) {
        size_t off = (size_t)row * N + col;
        float v = acc[t][r] + bv;
        if (accum) v += C[off];
        C[off] = v;
      }
    }
  }
}

// ---------------------------------------------------- RMSNorm (+ RoPE) per head
// grid = nrows * HEADS blocks of 128 threads; each block normalizes one 128-vector.
__global__ void rmsnorm_rope_kernel(const float* __restrict__ x, const float* __restrict__ w,
                                    const float* __restrict__ rc, const float* __restrict__ rs,
                                    __bf16* __restrict__ out, float eps, int do_rope) {
  const int row = blockIdx.x;
  const int token = row / HEADS;
  const int head = row % HEADS;
  const int d = threadIdx.x;
  const size_t base = (size_t)token * DIM + (size_t)head * HD;
  float v = x[base + d];
  float ss = v * v;
#pragma unroll
  for (int off = 16; off; off >>= 1) ss += __shfl_xor(ss, off, 32);
  __shared__ float part[4];
  if ((threadIdx.x & 31) == 0) part[threadIdx.x >> 5] = ss;
  __syncthreads();
  float tot = part[0] + part[1] + part[2] + part[3];
  float y = v * rsqrtf(tot * (1.0f / HD) + eps);
  if (w) y *= w[d];
  if (do_rope) {
    float mate = __shfl_xor(y, 1, 32);             // partner of the even/odd pair
    float rot = (d & 1) ? mate : -mate;            // rot[2i]=-x[2i+1], rot[2i+1]=x[2i]
    y = y * rc[(size_t)token * HD + d] + rot * rs[(size_t)token * HD + d];
  }
  out[base + d] = (__bf16)y;
}

// ---------------------------------------------------------------- attention
// One block = 64 q rows x 1 head. 128 threads = 4 waves, each wave owns 16 rows.
// KV tiles of 64 keys in LDS (V transposed on store); S = Q K^T (16 wmma/tile),
// online softmax (16-lane shfl reductions per C-layout row), P staged via LDS
// to A-layout, P@V (16 wmma/tile). 3 barriers per 64 keys.
__device__ __forceinline__ void load_q_frags(const __bf16* __restrict__ Qg, int q0, int head,
                                             __bf16* Qs, v16bf* qf,
                                             int tid, int wave, int half, int l16) {
  __syncthreads();
  for (int i = tid; i < 64 * 16; i += 128) {
    int r = i >> 4, c = (i & 15) * 8;
    *(uint4*)&Qs[r * 128 + c] = *(const uint4*)&Qg[(size_t)(q0 + r) * DIM + (size_t)head * HD + c];
  }
  __syncthreads();
#pragma unroll
  for (int c = 0; c < 4; ++c)
    qf[c] = *(const v16bf*)&Qs[(wave * 16 + l16) * 128 + c * 32 + half * 16];
}

__device__ __forceinline__ void attn_tile64(const __bf16* __restrict__ Kg,
                                            const __bf16* __restrict__ Vg,
                                            int key0, int head,
                                            __bf16* Ks, __bf16* Vt, __bf16* Psw,
                                            const v16bf* qf, v8f* acc,
                                            float* mrow, float* lrow,
                                            int tid, int half, int l16) {
  __syncthreads();  // previous tile consumed
  for (int i = tid; i < 64 * 16; i += 128) {
    int kk = i >> 4, c = (i & 15) * 8;
    const size_t g = (size_t)(key0 + kk) * DIM + (size_t)head * HD + c;
    *(uint4*)&Ks[kk * 128 + c] = *(const uint4*)&Kg[g];        // K: row major [key][d]
    const __bf16* vp = &Vg[g];
#pragma unroll
    for (int j = 0; j < 8; ++j) Vt[(c + j) * 64 + kk] = vp[j]; // V: transposed [d][key]
  }
  __syncthreads();

  v8f z8 = {0.f, 0.f, 0.f, 0.f, 0.f, 0.f, 0.f, 0.f};
  v8f s[4];
#pragma unroll
  for (int t = 0; t < 4; ++t) s[t] = z8;
#pragma unroll
  for (int c = 0; c < 4; ++c) {        // 4 chunks of head-dim 32
#pragma unroll
    for (int t = 0; t < 4; ++t) {      // 4 key n-tiles of 16
      v16bf b = *(const v16bf*)&Ks[(t * 16 + l16) * 128 + c * 32 + half * 16];
      s[t] = WMMA_BF16(qf[c], b, s[t]);
    }
  }

  float corr[8];
#pragma unroll
  for (int r = 0; r < 8; ++r) {        // row = r + 8*half
    float x0 = s[0][r] * SOFT_SCALE;
    float x1 = s[1][r] * SOFT_SCALE;
    float x2 = s[2][r] * SOFT_SCALE;
    float x3 = s[3][r] * SOFT_SCALE;
    float mx = fmaxf(fmaxf(x0, x1), fmaxf(x2, x3));
#pragma unroll
    for (int off = 1; off < 16; off <<= 1) mx = fmaxf(mx, __shfl_xor(mx, off, 32));
    float mnew = fmaxf(mrow[r], mx);
    float c = __expf(mrow[r] - mnew);
    float p0 = __expf(x0 - mnew);
    float p1 = __expf(x1 - mnew);
    float p2 = __expf(x2 - mnew);
    float p3 = __expf(x3 - mnew);
    float rsum = (p0 + p1) + (p2 + p3);
#pragma unroll
    for (int off = 1; off < 16; off <<= 1) rsum += __shfl_xor(rsum, off, 32);
    lrow[r] = lrow[r] * c + rsum;
    mrow[r] = mnew;
    corr[r] = c;
    int prow = r + half * 8;
    Psw[prow * 64 + l16]      = (__bf16)p0;
    Psw[prow * 64 + 16 + l16] = (__bf16)p1;
    Psw[prow * 64 + 32 + l16] = (__bf16)p2;
    Psw[prow * 64 + 48 + l16] = (__bf16)p3;
  }
#pragma unroll
  for (int t = 0; t < 8; ++t)
#pragma unroll
    for (int r = 0; r < 8; ++r) acc[t][r] *= corr[r];

  __syncthreads();  // P visible (also fences Ks/Vt for next tile)
  v16bf pf0 = *(const v16bf*)&Psw[l16 * 64 + half * 16];        // keys 0..31
  v16bf pf1 = *(const v16bf*)&Psw[l16 * 64 + 32 + half * 16];   // keys 32..63
#pragma unroll
  for (int t = 0; t < 8; ++t) {        // 8 head-dim tiles of 16
    v16bf vf0 = *(const v16bf*)&Vt[(t * 16 + l16) * 64 + half * 16];
    v16bf vf1 = *(const v16bf*)&Vt[(t * 16 + l16) * 64 + 32 + half * 16];
    acc[t] = WMMA_BF16(pf0, vf0, acc[t]);
    acc[t] = WMMA_BF16(pf1, vf1, acc[t]);
  }
}

__global__ void flash_attn_kernel(const __bf16* __restrict__ Qn, const __bf16* __restrict__ Kn,
                                  const __bf16* __restrict__ Vb, const __bf16* __restrict__ Qraw,
                                  const __bf16* __restrict__ IPK, const __bf16* __restrict__ IPV,
                                  float* __restrict__ out) {
  __shared__ __bf16 Qs[64 * 128];
  __shared__ __bf16 Ks[64 * 128];
  __shared__ __bf16 Vt[128 * 64];
  __shared__ __bf16 Ps[4][16 * 64];
  const int tid = threadIdx.x;
  const int wave = tid >> 5;
  const int lane = tid & 31;
  const int half = lane >> 4;
  const int l16 = lane & 15;
  const int head = blockIdx.y;
  const int q0 = blockIdx.x * 64;
  __bf16* Psw = &Ps[wave][0];

  v8f z8 = {0.f, 0.f, 0.f, 0.f, 0.f, 0.f, 0.f, 0.f};
  v16bf qf[4];
  v8f acc[8];
  float mrow[8], lrow[8];

  // ---- main attention: softmax over all SEQ keys (rows < BLK see zero bias)
#pragma unroll
  for (int t = 0; t < 8; ++t) acc[t] = z8;
#pragma unroll
  for (int r = 0; r < 8; ++r) { mrow[r] = -1e30f; lrow[r] = 0.f; }
  load_q_frags(Qn, q0, head, Qs, qf, tid, wave, half, l16);
  for (int k0 = 0; k0 < SEQ; k0 += 64)
    attn_tile64(Kn, Vb, k0, head, Ks, Vt, Psw, qf, acc, mrow, lrow, tid, half, l16);
  float inv[8];
#pragma unroll
  for (int r = 0; r < 8; ++r) inv[r] = 1.f / lrow[r];
#pragma unroll
  for (int t = 0; t < 8; ++t)
#pragma unroll
    for (int r = 0; r < 8; ++r) acc[t][r] *= inv[r];   // normalized main output

  // ---- IP attention: raw (pre-norm/rope) queries vs 64 image-derived keys
  v8f acc2[8];
#pragma unroll
  for (int t = 0; t < 8; ++t) acc2[t] = z8;
#pragma unroll
  for (int r = 0; r < 8; ++r) { mrow[r] = -1e30f; lrow[r] = 0.f; }
  load_q_frags(Qraw, q0, head, Qs, qf, tid, wave, half, l16);
  attn_tile64(IPK, IPV, 0, head, Ks, Vt, Psw, qf, acc2, mrow, lrow, tid, half, l16);
#pragma unroll
  for (int r = 0; r < 8; ++r) inv[r] = 1.f / lrow[r];

  // ---- fuse + store: out[token, head*HD + d]
#pragma unroll
  for (int t = 0; t < 8; ++t) {
    int d = t * 16 + l16;
#pragma unroll
    for (int r = 0; r < 8; ++r) {
      int row = q0 + wave * 16 + r + half * 8;
      out[(size_t)row * DIM + (size_t)head * HD + d] =
          acc[t][r] + IP_SCALE * (acc2[t][r] * inv[r]);
    }
  }
}

// ---------------------------------------------------------------- host side
static void launch_cast(const float* s, __bf16* d, long n, hipStream_t st) {
  long want = (n + 1023) / 1024;
  int blocks = (int)(want > 4096 ? 4096 : (want < 1 ? 1 : want));
  cast_f32_bf16<<<blocks, 256, 0, st>>>(s, d, n);
}

static void gemm_nt(const __bf16* A, const __bf16* W, const float* bias, float* C,
                    int M, int N, int K, int accum, hipStream_t st) {
  dim3 g((N + 127) / 128, (M + 63) / 64);
  wmma_gemm_nt<<<g, 256, 0, st>>>(A, W, bias, C, M, N, K, accum);
}

extern "C" void kernel_launch(void* const* d_in, const int* in_sizes, int n_in,
                              void* d_out, int out_size, void* d_ws, size_t ws_size,
                              hipStream_t stream) {
  (void)in_sizes; (void)n_in; (void)out_size; (void)ws_size;
  const float* hs   = (const float*)d_in[0];
  const float* img  = (const float*)d_in[1];
  const float* rc   = (const float*)d_in[2];
  const float* rs   = (const float*)d_in[3];
  const float* wq   = (const float*)d_in[4];
  const float* bq   = (const float*)d_in[5];
  const float* wk   = (const float*)d_in[6];
  const float* bk   = (const float*)d_in[7];
  const float* wv   = (const float*)d_in[8];
  const float* bv   = (const float*)d_in[9];
  const float* nqw  = (const float*)d_in[10];
  const float* nkw  = (const float*)d_in[11];
  const float* qd   = (const float*)d_in[12];
  const float* qu   = (const float*)d_in[13];
  const float* kd   = (const float*)d_in[14];
  const float* ku   = (const float*)d_in[15];
  const float* vd   = (const float*)d_in[16];
  const float* vu   = (const float*)d_in[17];
  const float* wkip = (const float*)d_in[18];
  const float* wvip = (const float*)d_in[19];

  char* p = (char*)d_ws;
  auto alloc = [&](size_t bytes) -> char* {
    char* r = p;
    p += (bytes + 255) & ~(size_t)255;
    return r;
  };
  const size_t SD = (size_t)SEQ * DIM;
  __bf16* hsb   = (__bf16*)alloc(SD * 2);
  __bf16* wqb   = (__bf16*)alloc((size_t)DIM * DIM * 2);
  __bf16* wkb   = (__bf16*)alloc((size_t)DIM * DIM * 2);
  __bf16* wvb   = (__bf16*)alloc((size_t)DIM * DIM * 2);
  __bf16* wkipb = (__bf16*)alloc((size_t)DIM * DIM * 2);
  __bf16* wvipb = (__bf16*)alloc((size_t)DIM * DIM * 2);
  __bf16* qdb   = (__bf16*)alloc((size_t)RANK * DIM * 2);
  __bf16* qub   = (__bf16*)alloc((size_t)DIM * RANK * 2);
  __bf16* kdb   = (__bf16*)alloc((size_t)RANK * DIM * 2);
  __bf16* kub   = (__bf16*)alloc((size_t)DIM * RANK * 2);
  __bf16* vdb   = (__bf16*)alloc((size_t)RANK * DIM * 2);
  __bf16* vub   = (__bf16*)alloc((size_t)DIM * RANK * 2);
  __bf16* imgb  = (__bf16*)alloc((size_t)NIP * DIM * 2);
  float*  qf32  = (float*)alloc(SD * 4);
  float*  kf32  = (float*)alloc(SD * 4);
  float*  vf32  = (float*)alloc(SD * 4);
  float*  tf32  = (float*)alloc((size_t)COND * RANK * 4);
  __bf16* tb    = (__bf16*)alloc((size_t)COND * RANK * 2);
  __bf16* qnb   = (__bf16*)alloc(SD * 2);
  __bf16* knb   = (__bf16*)alloc(SD * 2);
  __bf16* vbb   = (__bf16*)alloc(SD * 2);
  __bf16* qrawb = (__bf16*)alloc((size_t)BLK * DIM * 2);
  float*  ipkf  = (float*)alloc((size_t)NIP * DIM * 4);
  float*  ipvf  = (float*)alloc((size_t)NIP * DIM * 4);
  __bf16* ipkb  = (__bf16*)alloc((size_t)NIP * DIM * 2);
  __bf16* ipvb  = (__bf16*)alloc((size_t)NIP * DIM * 2);

  // 1) casts to bf16
  launch_cast(hs, hsb, (long)SD, stream);
  launch_cast(wq, wqb, (long)DIM * DIM, stream);
  launch_cast(wk, wkb, (long)DIM * DIM, stream);
  launch_cast(wv, wvb, (long)DIM * DIM, stream);
  launch_cast(wkip, wkipb, (long)DIM * DIM, stream);
  launch_cast(wvip, wvipb, (long)DIM * DIM, stream);
  launch_cast(qd, qdb, (long)RANK * DIM, stream);
  launch_cast(qu, qub, (long)DIM * RANK, stream);
  launch_cast(kd, kdb, (long)RANK * DIM, stream);
  launch_cast(ku, kub, (long)DIM * RANK, stream);
  launch_cast(vd, vdb, (long)RANK * DIM, stream);
  launch_cast(vu, vub, (long)DIM * RANK, stream);
  launch_cast(img, imgb, (long)NIP * DIM, stream);

  // 2) QKV projections (bias added in epilogue)
  gemm_nt(hsb, wqb, bq, qf32, SEQ, DIM, DIM, 0, stream);
  gemm_nt(hsb, wkb, bk, kf32, SEQ, DIM, DIM, 0, stream);
  gemm_nt(hsb, wvb, bv, vf32, SEQ, DIM, DIM, 0, stream);

  // 3) raw q (pre-LoRA/norm/rope) for the IP attention: rows [0, BLK)
  launch_cast(qf32, qrawb, (long)BLK * DIM, stream);

  // 4) LoRA: only masked rows [BLK, SEQ) contribute; down -> cast -> up(+=)
  const __bf16* hmb = hsb + (size_t)BLK * DIM;
  gemm_nt(hmb, qdb, nullptr, tf32, COND, RANK, DIM, 0, stream);
  launch_cast(tf32, tb, (long)COND * RANK, stream);
  gemm_nt(tb, qub, nullptr, qf32 + (size_t)BLK * DIM, COND, DIM, RANK, 1, stream);
  gemm_nt(hmb, kdb, nullptr, tf32, COND, RANK, DIM, 0, stream);
  launch_cast(tf32, tb, (long)COND * RANK, stream);
  gemm_nt(tb, kub, nullptr, kf32 + (size_t)BLK * DIM, COND, DIM, RANK, 1, stream);
  gemm_nt(hmb, vdb, nullptr, tf32, COND, RANK, DIM, 0, stream);
  launch_cast(tf32, tb, (long)COND * RANK, stream);
  gemm_nt(tb, vub, nullptr, vf32 + (size_t)BLK * DIM, COND, DIM, RANK, 1, stream);

  // 5) RMSNorm + RoPE on q/k; v just cast
  rmsnorm_rope_kernel<<<SEQ * HEADS, 128, 0, stream>>>(qf32, nqw, rc, rs, qnb, 1e-6f, 1);
  rmsnorm_rope_kernel<<<SEQ * HEADS, 128, 0, stream>>>(kf32, nkw, rc, rs, knb, 1e-6f, 1);
  launch_cast(vf32, vbb, (long)SD, stream);

  // 6) IP projections + norm
  gemm_nt(imgb, wkipb, nullptr, ipkf, NIP, DIM, DIM, 0, stream);
  gemm_nt(imgb, wvipb, nullptr, ipvf, NIP, DIM, DIM, 0, stream);
  rmsnorm_rope_kernel<<<NIP * HEADS, 128, 0, stream>>>(ipkf, nullptr, nullptr, nullptr, ipkb, 1e-5f, 0);
  launch_cast(ipvf, ipvb, (long)NIP * DIM, stream);

  // 7) fused main + IP attention -> d_out (BLK x DIM, fp32)
  dim3 ag(BLK / 64, HEADS);
  flash_attn_kernel<<<ag, 128, 0, stream>>>(qnb, knb, vbb, qrawb, ipkb, ipvb, (float*)d_out);
}
